// PointNetPlusPlusSemanticSeg_SSG_5549097746746
// MI455X (gfx1250) — matrix operand order repaired and checked by reading, not compiled, not run on hardware
//
#include <hip/hip_runtime.h>
#include <hip/hip_bf16.h>

typedef __attribute__((ext_vector_type(16))) _Float16 v16h;
typedef __attribute__((ext_vector_type(8)))  _Float16 v8h;
typedef __attribute__((ext_vector_type(8)))  float    v8f;

#define EPS_F 1e-8f

// ---------------------------------------------------------------------------
// (B, C, N) -> (B, N, C) transpose; also extracts xyz (first 3 channels)
// ---------------------------------------------------------------------------
__global__ __launch_bounds__(256)
void k_transpose(const float* __restrict__ src, float* __restrict__ pc,
                 float* __restrict__ xyz, int C, int N, int total) {
  int i = blockIdx.x * blockDim.x + threadIdx.x;
  if (i >= total) return;
  int b = i / N, n = i % N;
  for (int c = 0; c < C; ++c) {
    float v = src[((size_t)b * C + c) * N + n];
    pc[((size_t)b * N + n) * C + c] = v;
    if (c < 3) xyz[((size_t)b * N + n) * 3 + c] = v;
  }
}

// ---------------------------------------------------------------------------
// Farthest point sampling: one workgroup (256 threads) per batch.
// Register-resident running min distance, LDS argmax tree with
// first-index tie-break (matches jnp.argmax).
// ---------------------------------------------------------------------------
__global__ __launch_bounds__(256)
void k_fps(const float* __restrict__ xyz, int N, int S, int* __restrict__ out) {
  int b = blockIdx.x;
  int t = threadIdx.x;
  const float* X = xyz + (size_t)b * N * 3;
  float mind[16];
  int per = (N + 255) >> 8;
  for (int i = 0; i < per; ++i) mind[i] = 1e10f;

  __shared__ float sval[256];
  __shared__ int   sidx[256];
  __shared__ int   sfar;
  int far = 0;

  for (int it = 0; it < S; ++it) {
    if (t == 0) out[(size_t)b * S + it] = far;
    float cx = X[(size_t)far * 3 + 0];
    float cy = X[(size_t)far * 3 + 1];
    float cz = X[(size_t)far * 3 + 2];
    float bv = -1.f; int bi = 0x7fffffff;
    for (int i = 0; i < per; ++i) {
      int n = t + (i << 8);
      if (n < N) {
        float dx = X[(size_t)n * 3 + 0] - cx;
        float dy = X[(size_t)n * 3 + 1] - cy;
        float dz = X[(size_t)n * 3 + 2] - cz;
        float d = dx * dx + dy * dy + dz * dz;
        float m = fminf(mind[i], d);
        mind[i] = m;
        if (m > bv) { bv = m; bi = n; }   // strict > keeps lowest owned index
      }
    }
    sval[t] = bv; sidx[t] = bi;
    __syncthreads();
    for (int s2 = 128; s2 > 0; s2 >>= 1) {
      if (t < s2) {
        float ov = sval[t + s2]; int oi = sidx[t + s2];
        if (ov > sval[t] || (ov == sval[t] && oi < sidx[t])) { sval[t] = ov; sidx[t] = oi; }
      }
      __syncthreads();
    }
    if (t == 0) sfar = sidx[0];
    __syncthreads();
    far = sfar;
  }
}

__global__ __launch_bounds__(256)
void k_gather3(const float* __restrict__ xyz, const int* __restrict__ idx,
               float* __restrict__ out, int Nin, int S, int total) {
  int i = blockIdx.x * blockDim.x + threadIdx.x;
  if (i >= total) return;
  int b = i / S;
  int n = idx[i];
  out[(size_t)i * 3 + 0] = xyz[((size_t)b * Nin + n) * 3 + 0];
  out[(size_t)i * 3 + 1] = xyz[((size_t)b * Nin + n) * 3 + 1];
  out[(size_t)i * 3 + 2] = xyz[((size_t)b * Nin + n) * 3 + 2];
}

// ---------------------------------------------------------------------------
// Ball query (reference semantics: first 32 in-range indices in ascending
// index order, padded with the first in-range index).
// ---------------------------------------------------------------------------
__global__ __launch_bounds__(256)
void k_ball(const float* __restrict__ xyz, const float* __restrict__ nxyz,
            int* __restrict__ gidx, int Nin, int S, float r2, int total) {
  int i = blockIdx.x * blockDim.x + threadIdx.x;
  if (i >= total) return;
  int b = i / S;
  const float* X = xyz + (size_t)b * Nin * 3;
  float cx = nxyz[(size_t)i * 3 + 0];
  float cy = nxyz[(size_t)i * 3 + 1];
  float cz = nxyz[(size_t)i * 3 + 2];
  int* g = gidx + (size_t)i * 32;
  int cnt = 0, first = 0; bool have = false;
  for (int n = 0; n < Nin && cnt < 32; ++n) {
    float dx = X[(size_t)n * 3 + 0] - cx;
    float dy = X[(size_t)n * 3 + 1] - cy;
    float dz = X[(size_t)n * 3 + 2] - cz;
    float d = dx * dx + dy * dy + dz * dz;
    if (d <= r2) {
      if (!have) { first = n; have = true; }
      g[cnt++] = n;
    }
  }
  for (; cnt < 32; ++cnt) g[cnt] = first;
}

// ---------------------------------------------------------------------------
// Build grouped f16 GEMM operand: [xyz - center (3) | features (Cfeat)] padded
// with zeros to Kpad. Rows are (b, s, j) flattened.
// ---------------------------------------------------------------------------
__global__ __launch_bounds__(256)
void k_group(const float* __restrict__ xyz, const float* __restrict__ nxyz,
             const float* __restrict__ feat, int Cfeat, const int* __restrict__ gidx,
             _Float16* __restrict__ Xo, int Kpad, int Nin, int S, int total) {
  int row = blockIdx.x * blockDim.x + threadIdx.x;
  if (row >= total) return;
  int q = row >> 5;               // (b, s)
  int b = q / S;
  int n = gidx[row];
  const float* xn = xyz  + ((size_t)b * Nin + n) * 3;
  const float* ct = nxyz + (size_t)q * 3;
  const float* f  = feat + ((size_t)b * Nin + n) * Cfeat;
  _Float16* o = Xo + (size_t)row * Kpad;
  for (int c = 0; c < Kpad; ++c) {
    float v;
    if (c < 3)                 v = xn[c] - ct[c];
    else if (c - 3 < Cfeat)    v = f[c - 3];
    else                       v = 0.f;
    o[c] = (_Float16)v;
  }
}

// ---------------------------------------------------------------------------
// Weight prep: f32 W[Cout][Cin] -> f16 Wp[Npad][Kpad] zero-padded. Keeping the
// reference (Cout, Cin) row-major layout makes each WMMA B-operand lane load a
// single contiguous 32B chunk.
// ---------------------------------------------------------------------------
__global__ __launch_bounds__(256)
void k_prep_w(const float* __restrict__ W, _Float16* __restrict__ Wp,
              int Cout, int Cin, int Npad, int Kpad) {
  int i = blockIdx.x * blockDim.x + threadIdx.x;
  if (i >= Npad * Kpad) return;
  int n = i / Kpad, k = i % Kpad;
  float v = (n < Cout && k < Cin) ? W[(size_t)n * Cin + k] : 0.f;
  Wp[i] = (_Float16)v;
}

// ---------------------------------------------------------------------------
// Generic fused GEMM: Y = act((X @ W^T + b) * gamma + beta)
//   X: f16 [M][Kpad] row-major, Wp: f16 [Npad][Kpad] row-major.
//   One wave computes a 16x16 D tile via v_wmma_f32_16x16x32_f16.
//   Dual output: f16 (chained layer input) and/or f32.
// C/D VGPR layout per CDNA5 ISA 7.12.2 (wave32); A/B use contiguous K (k = 16h+s).
// ---------------------------------------------------------------------------
__global__ __launch_bounds__(256)
void k_gemm_wmma(const _Float16* __restrict__ X, const _Float16* __restrict__ Wp,
                 const float* __restrict__ bias, const float* __restrict__ gamma,
                 const float* __restrict__ beta,
                 int M, int Kpad, int Cout, int relu,
                 _Float16* __restrict__ o16, float* __restrict__ o32, int ldo) {
  const int lane = threadIdx.x & 31;
  const int wave = threadIdx.x >> 5;
  const int tm = blockIdx.x * 128 + wave * 16;
  const int tn = blockIdx.y * 16;
  if (tm >= M) return;                    // wave-uniform: EXEC stays all-ones
  const int hh  = lane >> 4;
  const int l15 = lane & 15;
  const int row  = tm + l15;              // A-operand row for this lane
  const int ncol = tn + l15;              // B-operand column for this lane
  const _Float16* xr = X  + (size_t)row  * Kpad;
  const _Float16* wr = Wp + (size_t)ncol * Kpad;

  v8f acc = {0.f, 0.f, 0.f, 0.f, 0.f, 0.f, 0.f, 0.f};
  for (int kk = 0; kk < Kpad; kk += 32) {
    // A 16x32 f16: lane holds 16 contiguous K values kk+16*hh+[0..15] of row
    v16h A = *(const v16h*)(xr + kk + 16 * hh);
    // B 32x16 f16: lane holds 16 contiguous K values of column ncol
    v16h Bm = *(const v16h*)(wr + kk + 16 * hh);
    acc = __builtin_amdgcn_wmma_f32_16x16x32_f16(false, A, false, Bm,
                                                 (short)0, acc, false, false);
  }

  float bb = (ncol < Cout) ? bias[ncol] : 0.f;
  float gg = 1.f, ee = 0.f;
  if (gamma) {
    gg = (ncol < Cout) ? gamma[ncol] : 1.f;
    ee = (ncol < Cout) ? beta[ncol]  : 0.f;
  }
#pragma unroll
  for (int r = 0; r < 8; ++r) {
    int m = tm + r + 8 * hh;              // C/D layout: VGPR r -> row r (+8 for hi lanes)
    float v = (acc[r] + bb) * gg + ee;
    if (relu) v = fmaxf(v, 0.f);
    if (ncol >= Cout) v = 0.f;
    if (o16) o16[(size_t)m * ldo + ncol] = (_Float16)v;
    if (o32) o32[(size_t)m * ldo + ncol] = v;
  }
}

// ---------------------------------------------------------------------------
// Max over the 32-sample axis; f16 in, f32 out.
// ---------------------------------------------------------------------------
__global__ __launch_bounds__(256)
void k_maxpool(const _Float16* __restrict__ Y, int ld, float* __restrict__ out,
               int Cout, int total) {
  int i = blockIdx.x * blockDim.x + threadIdx.x;
  if (i >= total) return;
  int c = i % Cout;
  int q = i / Cout;
  float m = -1e30f;
  for (int j = 0; j < 32; ++j)
    m = fmaxf(m, (float)Y[((size_t)q * 32 + j) * ld + c]);
  out[i] = m;
}

// ---------------------------------------------------------------------------
// Feature propagation: 3-NN inverse-distance interpolation, concat with
// skip features, write f16 GEMM operand directly (zero-padded to Kpad).
// ---------------------------------------------------------------------------
__global__ __launch_bounds__(256)
void k_interp(const float* __restrict__ xyz1, int N1,
              const float* __restrict__ xyz2, int S2,
              const float* __restrict__ p1, int C1,
              const float* __restrict__ p2, int C2,
              _Float16* __restrict__ Xo, int Kpad, int total) {
  int i = blockIdx.x * blockDim.x + threadIdx.x;
  if (i >= total) return;
  int b = i / N1;
  const float* a  = xyz1 + (size_t)i * 3;
  const float* X2 = xyz2 + (size_t)b * S2 * 3;
  float ax = a[0], ay = a[1], az = a[2];
  float d0 = 1e30f, d1 = 1e30f, d2 = 1e30f;
  int   i0 = 0, i1 = 0, i2 = 0;
  for (int s = 0; s < S2; ++s) {
    float dx = X2[(size_t)s * 3 + 0] - ax;
    float dy = X2[(size_t)s * 3 + 1] - ay;
    float dz = X2[(size_t)s * 3 + 2] - az;
    float d = dx * dx + dy * dy + dz * dz;
    if (d < d0)      { d2 = d1; i2 = i1; d1 = d0; i1 = i0; d0 = d; i0 = s; }
    else if (d < d1) { d2 = d1; i2 = i1; d1 = d;  i1 = s; }
    else if (d < d2) { d2 = d;  i2 = s; }
  }
  float w0 = 1.f / (d0 + EPS_F);
  float w1 = 1.f / (d1 + EPS_F);
  float w2 = 1.f / (d2 + EPS_F);
  float ws = w0 + w1 + w2;
  w0 /= ws; w1 /= ws; w2 /= ws;

  _Float16* o = Xo + (size_t)i * Kpad;
  for (int c = 0; c < C1; ++c)
    o[c] = (_Float16)p1[(size_t)i * C1 + c];
  const float* q0 = p2 + ((size_t)b * S2 + i0) * C2;
  const float* q1 = p2 + ((size_t)b * S2 + i1) * C2;
  const float* q2 = p2 + ((size_t)b * S2 + i2) * C2;
  for (int k = 0; k < C2; ++k)
    o[C1 + k] = (_Float16)(w0 * q0[k] + w1 * q1[k] + w2 * q2[k]);
  for (int c = C1 + C2; c < Kpad; ++c)
    o[c] = (_Float16)0.f;
}

// ---------------------------------------------------------------------------
// log_softmax over the 13 classes (logits stored with leading dim 16)
// ---------------------------------------------------------------------------
__global__ __launch_bounds__(256)
void k_logsoftmax(const float* __restrict__ logits, int ld, float* __restrict__ out,
                  int total, int K) {
  int i = blockIdx.x * blockDim.x + threadIdx.x;
  if (i >= total) return;
  const float* L = logits + (size_t)i * ld;
  float m = -1e30f;
  for (int k = 0; k < K; ++k) m = fmaxf(m, L[k]);
  float s = 0.f;
  for (int k = 0; k < K; ++k) s += __expf(L[k] - m);
  float lse = __logf(s);
  for (int k = 0; k < K; ++k) out[(size_t)i * K + k] = L[k] - m - lse;
}

// p4 (B, 16, 512) -> out (B, 512, 16)
__global__ __launch_bounds__(256)
void k_out_p4(const float* __restrict__ p4, float* __restrict__ out, int total) {
  int i = blockIdx.x * blockDim.x + threadIdx.x;
  if (i >= total) return;
  int b = i / (512 * 16);
  int r = i % (512 * 16);
  int c = r / 16;
  int s = r % 16;
  out[i] = p4[((size_t)b * 16 + s) * 512 + c];
}

// ---------------------------------------------------------------------------
// Host orchestration
// ---------------------------------------------------------------------------
extern "C" void kernel_launch(void* const* d_in, const int* in_sizes, int n_in,
                              void* d_out, int out_size, void* d_ws, size_t ws_size,
                              hipStream_t stream) {
  (void)in_sizes; (void)n_in; (void)out_size; (void)ws_size;
  const int B = 16, N = 4096, CIN = 9, NCLS = 13;

  char* base = (char*)d_ws;
  size_t off = 0;
  auto alloc = [&](size_t bytes) -> char* {
    off = (off + 255) & ~(size_t)255;
    char* p = base + off;
    off += bytes;
    return p;
  };

  float* pc     = (float*)alloc((size_t)B * N * 9 * 4);
  float* xyz0   = (float*)alloc((size_t)B * N * 3 * 4);
  float* xyz1   = (float*)alloc((size_t)B * 1024 * 3 * 4);
  float* xyz2   = (float*)alloc((size_t)B * 256 * 3 * 4);
  float* xyz3   = (float*)alloc((size_t)B * 64 * 3 * 4);
  float* xyz4   = (float*)alloc((size_t)B * 16 * 3 * 4);
  int*   fpsidx = (int*)alloc((size_t)B * 1024 * 4);
  int*   gidx   = (int*)alloc((size_t)B * 1024 * 32 * 4);
  float* p1     = (float*)alloc((size_t)B * 1024 * 64 * 4);
  float* p2     = (float*)alloc((size_t)B * 256 * 128 * 4);
  float* p3     = (float*)alloc((size_t)B * 64 * 256 * 4);
  float* p4     = (float*)alloc((size_t)B * 16 * 512 * 4);
  float* u1     = (float*)alloc((size_t)B * 64 * 256 * 4);
  float* u2     = (float*)alloc((size_t)B * 256 * 256 * 4);
  float* u3     = (float*)alloc((size_t)B * 1024 * 128 * 4);
  float* logits = (float*)alloc((size_t)B * N * 16 * 4);
  _Float16* bufA = (_Float16*)alloc((size_t)36 * 1024 * 1024);  // max 524288x32 f16
  _Float16* bufB = (_Float16*)alloc((size_t)68 * 1024 * 1024);  // max 524288x64 f16

  // Layer table: {weight input idx, Cin, Cout, Kpad, Npad}
  struct GL { int widx, Cin, Cout, Kpad, Npad; };
  const GL L[23] = {
    {1, 12, 32, 32, 32},   {5, 32, 32, 32, 32},   {9, 32, 64, 32, 64},     // sa1
    {13, 67, 64, 96, 64},  {17, 64, 64, 64, 64},  {21, 64, 128, 64, 128},  // sa2
    {25, 131, 128, 160, 128}, {29, 128, 128, 128, 128}, {33, 128, 256, 128, 256}, // sa3
    {37, 259, 256, 288, 256}, {41, 256, 256, 256, 256}, {45, 256, 512, 256, 512}, // sa4
    {49, 768, 256, 768, 256}, {53, 256, 256, 256, 256},                    // fp1
    {57, 384, 256, 384, 256}, {61, 256, 256, 256, 256},                    // fp2
    {65, 320, 256, 320, 256}, {69, 256, 128, 256, 128},                    // fp3
    {73, 128, 128, 128, 128}, {77, 128, 128, 128, 128}, {81, 128, 128, 128, 128}, // fp4
    {85, 128, 128, 128, 128},                                              // cls1
    {89, 128, 13, 128, 16}                                                 // final
  };

  _Float16* Wp[23];
  for (int i = 0; i < 23; ++i) {
    size_t ne = (size_t)L[i].Npad * L[i].Kpad;
    Wp[i] = (_Float16*)alloc(ne * 2);
    int blocks = (int)((ne + 255) / 256);
    k_prep_w<<<blocks, 256, 0, stream>>>((const float*)d_in[L[i].widx], Wp[i],
                                         L[i].Cout, L[i].Cin, L[i].Npad, L[i].Kpad);
  }

  auto gemm = [&](int li, const _Float16* Xi, int M, _Float16* o16, float* o32, int relu) {
    const GL& g = L[li];
    const float* bias  = (const float*)d_in[g.widx + 1];
    const float* gam   = (li == 22) ? nullptr : (const float*)d_in[g.widx + 2];
    const float* bet   = (li == 22) ? nullptr : (const float*)d_in[g.widx + 3];
    dim3 grid((M + 127) / 128, g.Npad / 16);
    k_gemm_wmma<<<grid, 256, 0, stream>>>(Xi, Wp[li], bias, gam, bet,
                                          M, g.Kpad, g.Cout, relu, o16, o32, g.Npad);
  };

  // pc / xyz0
  { int t = B * N;
    k_transpose<<<(t + 255) / 256, 256, 0, stream>>>((const float*)d_in[0], pc, xyz0, CIN, N, t); }

  auto run_sa = [&](const float* xyzin, int Nin, const float* featin, int Cfeat,
                    int S, float radius, int li0, float* xyzout, float* pooled) {
    k_fps<<<B, 256, 0, stream>>>(xyzin, Nin, S, fpsidx);
    int tq = B * S;
    k_gather3<<<(tq + 255) / 256, 256, 0, stream>>>(xyzin, fpsidx, xyzout, Nin, S, tq);
    k_ball<<<(tq + 255) / 256, 256, 0, stream>>>(xyzin, xyzout, gidx, Nin, S,
                                                 radius * radius, tq);
    int M = B * S * 32;
    k_group<<<(M + 255) / 256, 256, 0, stream>>>(xyzin, xyzout, featin, Cfeat, gidx,
                                                 bufA, L[li0].Kpad, Nin, S, M);
    gemm(li0 + 0, bufA, M, bufB, nullptr, 1);
    gemm(li0 + 1, bufB, M, bufA, nullptr, 1);
    gemm(li0 + 2, bufA, M, bufB, nullptr, 1);
    int Cout = L[li0 + 2].Cout, ld = L[li0 + 2].Npad;
    int tp = B * S * Cout;
    k_maxpool<<<(tp + 255) / 256, 256, 0, stream>>>(bufB, ld, pooled, Cout, tp);
  };

  run_sa(xyz0, N,    pc, 9,   1024, 0.1f, 0, xyz1, p1);
  run_sa(xyz1, 1024, p1, 64,  256,  0.2f, 3, xyz2, p2);
  run_sa(xyz2, 256,  p2, 128, 64,   0.4f, 6, xyz3, p3);
  run_sa(xyz3, 64,   p3, 256, 16,   0.8f, 9, xyz4, p4);

  auto run_fp2 = [&](const float* xa, int N1, const float* xb, int S2,
                     const float* pa, int C1, const float* pb, int C2,
                     int li0, float* uout) {
    int M = B * N1;
    k_interp<<<(M + 255) / 256, 256, 0, stream>>>(xa, N1, xb, S2, pa, C1, pb, C2,
                                                  bufA, L[li0].Kpad, M);
    gemm(li0 + 0, bufA, M, bufB, nullptr, 1);
    gemm(li0 + 1, bufB, M, bufA, uout, 1);
  };
  run_fp2(xyz3, 64,   xyz4, 16,  p3, 256, p4, 512, 12, u1);
  run_fp2(xyz2, 256,  xyz3, 64,  p2, 128, u1, 256, 14, u2);
  run_fp2(xyz1, 1024, xyz2, 256, p1, 64,  u2, 256, 16, u3);

  // fp4 (no skip features) + classification head
  {
    int M = B * N;
    k_interp<<<(M + 255) / 256, 256, 0, stream>>>(xyz0, N, xyz1, 1024,
                                                  nullptr, 0, u3, 128,
                                                  bufA, 128, M);
    gemm(18, bufA, M, bufB, nullptr, 1);
    gemm(19, bufB, M, bufA, nullptr, 1);
    gemm(20, bufA, M, bufB, nullptr, 1);
    gemm(21, bufB, M, bufA, nullptr, 1);      // cls hidden layer
    gemm(22, bufA, M, nullptr, logits, 0);    // final linear (no relu / no bn)

    int t = B * N;
    k_logsoftmax<<<(t + 255) / 256, 256, 0, stream>>>(logits, 16, (float*)d_out, t, NCLS);
    int t2 = B * 512 * 16;
    k_out_p4<<<(t2 + 255) / 256, 256, 0, stream>>>(p4, (float*)d_out + (size_t)B * N * NCLS, t2);
  }
}
